// MoChA_25039659335887
// MI455X (gfx1250) — matrix operand order repaired and checked by reading, not compiled, not run on hardware
//
#include <hip/hip_runtime.h>
#include <hip/hip_bf16.h>
#include <math.h>

#define B 16
#define T 8192
#define D 512
#define A 512
#define CHUNK 4
#define NEG_INF (-3.402823466e+38f)
#define EPSV 1e-8f

typedef __attribute__((ext_vector_type(16))) __bf16        v16bf;
typedef __attribute__((ext_vector_type(2)))  __bf16        v2bf;
typedef __attribute__((ext_vector_type(8)))  float         v8f;
typedef __attribute__((ext_vector_type(4)))  unsigned int  v4u;
typedef __attribute__((ext_vector_type(8)))  unsigned int  v8u;

__device__ __forceinline__ float fast_tanh(float x) {
#if __has_builtin(__builtin_amdgcn_tanhf)
    return __builtin_amdgcn_tanhf(x);    // v_tanh_f32
#else
    return tanhf(x);
#endif
}

__device__ __forceinline__ v16bf mk16(v4u lo, v4u hi) {
    v8u t;
    t[0] = lo[0]; t[1] = lo[1]; t[2] = lo[2]; t[3] = lo[3];
    t[4] = hi[0]; t[5] = hi[1]; t[6] = hi[2]; t[7] = hi[3];
    return __builtin_bit_cast(v16bf, t);
}

// ---------------------------------------------------------------------------
// Prep kernel 1: v_w = vg / ||vv|| * vv   (for both mono and chunk)
// ---------------------------------------------------------------------------
__global__ __launch_bounds__(256) void vw_kernel(const float* __restrict__ mvv,
                                                 const float* __restrict__ mvg,
                                                 const float* __restrict__ cvv,
                                                 const float* __restrict__ cvg,
                                                 float* __restrict__ vwm,
                                                 float* __restrict__ vwc) {
    __shared__ float sb[256];
    const float* vv = blockIdx.x ? cvv : mvv;
    const float* vg = blockIdx.x ? cvg : mvg;
    float*      out = blockIdx.x ? vwc : vwm;
    int tid = threadIdx.x;
    float s = 0.f;
    for (int i = tid; i < A; i += 256) { float x = vv[i]; s += x * x; }
    sb[tid] = s; __syncthreads();
    for (int off = 128; off >= 1; off >>= 1) {
        if (tid < off) sb[tid] += sb[tid + off];
        __syncthreads();
    }
    float scale = vg[0] / sqrtf(sb[0]);
    for (int i = tid; i < A; i += 256) out[i] = scale * vv[i];
}

// ---------------------------------------------------------------------------
// Prep kernel 2: qb[b,a] = sum_d query[b,d]*wq[d,a] + bk[a]
// ---------------------------------------------------------------------------
__global__ __launch_bounds__(512) void qb_kernel(const float* __restrict__ query,
                                                 const float* __restrict__ mwq,
                                                 const float* __restrict__ mbk,
                                                 const float* __restrict__ cwq,
                                                 const float* __restrict__ cbk,
                                                 float* __restrict__ qbm,
                                                 float* __restrict__ qbc) {
    int b = blockIdx.x;
    int a = threadIdx.x;
    const float* wq = blockIdx.y ? cwq : mwq;
    const float* bk = blockIdx.y ? cbk : mbk;
    float*      out = blockIdx.y ? qbc : qbm;
    const float* q  = query + (size_t)b * D;
    float s = 0.f;
    for (int d = 0; d < D; ++d) s += q[d] * wq[(size_t)d * A + a];
    out[b * A + a] = s + bk[a];
}

// ---------------------------------------------------------------------------
// Prep kernel 3: wkT_bf16[a*D + d] = bf16(wk[d*A + a])  (both matrices)
// ---------------------------------------------------------------------------
__global__ __launch_bounds__(256) void wkt_kernel(const float* __restrict__ mwk,
                                                  const float* __restrict__ cwk,
                                                  __bf16* __restrict__ wtm,
                                                  __bf16* __restrict__ wtc) {
    size_t idx = (size_t)blockIdx.x * 256 + threadIdx.x;
    const float* wk = blockIdx.y ? cwk : mwk;
    __bf16*     out = blockIdx.y ? wtc : wtm;
    int a = (int)(idx / D);
    int d = (int)(idx % D);
    out[(size_t)a * D + d] = (__bf16)wk[(size_t)d * A + a];
}

// ---------------------------------------------------------------------------
// Main energy kernel. One wave owns a 16-row t-tile:
//   phase 1: key tile (16x512 f32) converted once -> LDS bf16 (16 KB/wave)
//   phase 2: 32 a-tiles x 16 k-chunks of v_wmma_f32_16x16x32_bf16 (x2 weights)
//            A frag = 2x ds_load_b128, B frag = 2x global_load_b128 per matrix
//   epilogue: e += v_tanh_f32(acc + q_bias) * v_w, shfl_xor N-reduction
// ---------------------------------------------------------------------------
#define EWAVES 4
__global__ __launch_bounds__(32 * EWAVES) void energy_kernel(
    const float*  __restrict__ key,
    const __bf16* __restrict__ wtm,
    const __bf16* __restrict__ wtc,
    const float*  __restrict__ qbm,
    const float*  __restrict__ qbc,
    const float*  __restrict__ vwm,
    const float*  __restrict__ vwc,
    const unsigned char* __restrict__ mask,
    const float* __restrict__ m_vb, const float* __restrict__ m_r,
    const float* __restrict__ c_vb, const float* __restrict__ c_r,
    float* __restrict__ e_mono,
    float* __restrict__ e_chunk) {

    __shared__ __align__(16) __bf16 ksh[EWAVES][16 * D];   // 64 KB total

    const int lane = threadIdx.x & 31;
    const int wave = threadIdx.x >> 5;
    const int b    = blockIdx.y;
    const int t0   = (blockIdx.x * EWAVES + wave) * 16;
    const int half = lane >> 4;      // 0: lanes 0-15, 1: lanes 16-31
    const int n    = lane & 15;      // A-frag row == C-frag column

    // ---- Phase 1: key tile f32 -> bf16 in LDS (rows are contiguous) ----
    {
        const float* kbase = key + ((size_t)b * T + t0) * D;  // 16*512 flat
        __bf16* dst = &ksh[wave][0];
        for (int j = 0; j < (16 * D) / 64; ++j) {             // 128 iters
            int e = 2 * (lane + j * 32);
            float2 kv = *(const float2*)(kbase + e);
            v2bf hp;
            hp[0] = (__bf16)kv.x;
            hp[1] = (__bf16)kv.y;
            *(v2bf*)(dst + e) = hp;                           // ds_store_b32
        }
    }
    __syncthreads();

    float eM[8], eC[8];
#pragma unroll
    for (int r = 0; r < 8; ++r) { eM[r] = 0.f; eC[r] = 0.f; }

    const __bf16* arow = &ksh[wave][0] + n * D + 8 * half;

    for (int a0 = 0; a0 < A; a0 += 16) {
        v8f accM, accC;
#pragma unroll
        for (int i = 0; i < 8; ++i) { accM[i] = 0.f; accC[i] = 0.f; }

        const __bf16* bmrow = wtm + (size_t)(a0 + n) * D + 8 * half;
        const __bf16* bcrow = wtc + (size_t)(a0 + n) * D + 8 * half;

        for (int d0 = 0; d0 < D; d0 += 32) {
            // A 16x32 bf16 (ISA 7.12.2): VGPR0-3 K=d0+8h+0..7, VGPR4-7 K=+16
            v4u alo = *(const v4u*)(arow + d0);        // ds_load_b128
            v4u ahi = *(const v4u*)(arow + d0 + 16);   // ds_load_b128
            // B 32x16 bf16: VGPR0-3 K=d0+8h+0..7, VGPR4-7 K=+16
            v4u mlo = *(const v4u*)(bmrow + d0);       // global_load_b128
            v4u mhi = *(const v4u*)(bmrow + d0 + 16);
            v4u clo = *(const v4u*)(bcrow + d0);
            v4u chi = *(const v4u*)(bcrow + d0 + 16);

            v16bf afrag = mk16(alo, ahi);
            v16bf bm    = mk16(mlo, mhi);
            v16bf bc    = mk16(clo, chi);

            accM = __builtin_amdgcn_wmma_f32_16x16x32_bf16(
                false, afrag, false, bm, (short)0, accM, false, false);
            accC = __builtin_amdgcn_wmma_f32_16x16x32_bf16(
                false, afrag, false, bc, (short)0, accC, false, false);
        }

        // Fused epilogue: tanh(k + q_bias) * v_w, partial-summed over a-tile.
        float qm = qbm[b * A + a0 + n], vm = vwm[a0 + n];
        float qc = qbc[b * A + a0 + n], vc = vwc[a0 + n];
#pragma unroll
        for (int r = 0; r < 8; ++r) {
            eM[r] += fast_tanh(accM[r] + qm) * vm;
            eC[r] += fast_tanh(accC[r] + qc) * vc;
        }
    }

    // Reduce the 16-lane N dimension (C layout: lane holds M = r + 8*half, N = n).
#pragma unroll
    for (int r = 0; r < 8; ++r) {
        for (int off = 8; off >= 1; off >>= 1) {
            eM[r] += __shfl_xor(eM[r], off, 16);
            eC[r] += __shfl_xor(eC[r], off, 16);
        }
    }

    float offM = m_vb[0] + m_r[0];
    float offC = c_vb[0] + c_r[0];
    if (n == 0) {
#pragma unroll
        for (int r = 0; r < 8; ++r) {
            int row = t0 + r + 8 * half;
            bool mk = mask[(size_t)b * T + row] != 0;
            e_mono [(size_t)b * T + row] = mk ? (eM[r] + offM) : NEG_INF;
            e_chunk[(size_t)b * T + row] = mk ? (eC[r] + offC) : NEG_INF;
        }
    }
}

// ---------------------------------------------------------------------------
// Per-batch scans: p, cumsum(log1mp), cumprod, cumsum(aw/cp), alpha,
// plus e_chunk max and softmax numerator (sexp).
// ---------------------------------------------------------------------------
__device__ __forceinline__ float block_scan_incl(float v, float* sb) {
    int tid = threadIdx.x;
    sb[tid] = v; __syncthreads();
    for (int off = 1; off < 256; off <<= 1) {
        float t = (tid >= off) ? sb[tid - off] : 0.f;
        __syncthreads();
        sb[tid] += t;
        __syncthreads();
    }
    return sb[tid];
}

__global__ __launch_bounds__(256) void scan_kernel(const float* __restrict__ e_mono,
                                                   const float* __restrict__ e_chunk,
                                                   const float* __restrict__ noise,
                                                   const float* __restrict__ aw_prev,
                                                   float* __restrict__ alpha_out,
                                                   float* __restrict__ sexp_out) {
    __shared__ float sb[256];
    int b = blockIdx.x, tid = threadIdx.x;

    // max over e_chunk[b,:]
    float mx = NEG_INF;
    for (int i = tid; i < T; i += 256) mx = fmaxf(mx, e_chunk[(size_t)b * T + i]);
    sb[tid] = mx; __syncthreads();
    for (int off = 128; off >= 1; off >>= 1) {
        if (tid < off) sb[tid] = fmaxf(sb[tid], sb[tid + off]);
        __syncthreads();
    }
    float maxE = sb[0]; __syncthreads();

    float carryL = 0.f, carryA = 0.f;
    for (int c = 0; c < T; c += 256) {
        size_t i = (size_t)b * T + c + tid;
        float em = e_mono[i];
        float p  = 1.f / (1.f + expf(-(em + noise[i])));
        float l  = logf(fminf(fmaxf(1.f - p, EPSV), 1.f));

        float inclL = block_scan_incl(l, sb);
        float totL  = sb[255];
        __syncthreads();
        float cp = expf(carryL + inclL - l);   // exclusive cumsum -> cumprod(1-p)

        float a2 = aw_prev[i] / fminf(fmaxf(cp, EPSV), 1.f);
        float inclA = block_scan_incl(a2, sb);
        float totA  = sb[255];
        __syncthreads();

        alpha_out[i] = p * cp * (carryA + inclA);

        float se = fmaxf(expf(e_chunk[i] - maxE), 1e-5f);
        sexp_out[i] = se;

        carryL += totL;
        carryA += totA;
    }
}

// ---------------------------------------------------------------------------
// beta[t] = sexp[t] * sum_{j=t..t+3} alpha[j] / (sum_{i=j-3..j} sexp[i])
// ---------------------------------------------------------------------------
__global__ __launch_bounds__(256) void beta_kernel(const float* __restrict__ alpha,
                                                   const float* __restrict__ sexp,
                                                   float* __restrict__ beta_out) {
    int idx = blockIdx.x * 256 + threadIdx.x;   // grid sized exactly B*T
    int b = idx / T, t = idx % T;
    const float* se = sexp  + (size_t)b * T;
    const float* al = alpha + (size_t)b * T;
    float s = 0.f;
    int jmax = (t + CHUNK - 1 < T - 1) ? (t + CHUNK - 1) : (T - 1);
    for (int j = t; j <= jmax; ++j) {
        float den = 0.f;
        int i0 = (j - (CHUNK - 1) > 0) ? (j - (CHUNK - 1)) : 0;
        for (int i = i0; i <= j; ++i) den += se[i];
        s += al[j] / den;
    }
    beta_out[idx] = se[t] * s;
}

// ---------------------------------------------------------------------------
// cv[b,d] = sum_t beta[b,t] * value[b,t,d]   (beta staged through LDS)
// ---------------------------------------------------------------------------
__global__ __launch_bounds__(256) void cv_kernel(const float* __restrict__ value,
                                                 const float* __restrict__ beta,
                                                 float* __restrict__ cv) {
    __shared__ float bsh[256];
    int b = blockIdx.y;
    int d = blockIdx.x * 256 + threadIdx.x;
    float acc = 0.f;
    for (int t0 = 0; t0 < T; t0 += 256) {
        __syncthreads();
        bsh[threadIdx.x] = beta[(size_t)b * T + t0 + threadIdx.x];
        __syncthreads();
        for (int tt = 0; tt < 256; ++tt)
            acc += bsh[tt] * value[((size_t)b * T + t0 + tt) * D + d];
    }
    cv[b * D + d] = acc;
}

// ---------------------------------------------------------------------------
extern "C" void kernel_launch(void* const* d_in, const int* in_sizes, int n_in,
                              void* d_out, int out_size, void* d_ws, size_t ws_size,
                              hipStream_t stream) {
    (void)in_sizes; (void)n_in; (void)out_size; (void)ws_size;

    const float* key     = (const float*)d_in[0];
    const float* value   = (const float*)d_in[1];
    const float* query   = (const float*)d_in[2];
    const float* noise   = (const float*)d_in[3];
    const float* aw_prev = (const float*)d_in[4];
    const unsigned char* mask = (const unsigned char*)d_in[5];
    const float* m_wk = (const float*)d_in[6];
    const float* m_bk = (const float*)d_in[7];
    const float* m_wq = (const float*)d_in[8];
    const float* m_vv = (const float*)d_in[9];
    const float* m_vg = (const float*)d_in[10];
    const float* m_vb = (const float*)d_in[11];
    const float* m_r  = (const float*)d_in[12];
    const float* c_wk = (const float*)d_in[13];
    const float* c_bk = (const float*)d_in[14];
    const float* c_wq = (const float*)d_in[15];
    const float* c_vv = (const float*)d_in[16];
    const float* c_vg = (const float*)d_in[17];
    const float* c_vb = (const float*)d_in[18];
    const float* c_r  = (const float*)d_in[19];

    // Outputs: (cv [B,1,D], alpha [B,T], beta [B,T]) concatenated flat.
    float* out_cv    = (float*)d_out;
    float* out_alpha = out_cv + B * D;
    float* out_beta  = out_alpha + B * T;

    // Workspace carve-up (~2.7 MB).
    float* ws       = (float*)d_ws;
    float* e_mono   = ws;                 ws += B * T;
    float* e_chunk  = ws;                 ws += B * T;
    float* sexp     = ws;                 ws += B * T;
    float* qbm      = ws;                 ws += B * A;
    float* qbc      = ws;                 ws += B * A;
    float* vwm      = ws;                 ws += A;
    float* vwc      = ws;                 ws += A;
    __bf16* wtm     = (__bf16*)ws;
    __bf16* wtc     = wtm + (size_t)A * D;

    vw_kernel<<<2, 256, 0, stream>>>(m_vv, m_vg, c_vv, c_vg, vwm, vwc);
    qb_kernel<<<dim3(B, 2), 512, 0, stream>>>(query, m_wq, m_bk, c_wq, c_bk, qbm, qbc);
    wkt_kernel<<<dim3((D * A) / 256, 2), 256, 0, stream>>>(m_wk, c_wk, wtm, wtc);

    energy_kernel<<<dim3(T / (16 * EWAVES), B), 32 * EWAVES, 0, stream>>>(
        key, wtm, wtc, qbm, qbc, vwm, vwc, mask,
        m_vb, m_r, c_vb, c_r, e_mono, e_chunk);

    scan_kernel<<<B, 256, 0, stream>>>(e_mono, e_chunk, noise, aw_prev,
                                       out_alpha, sexp);
    beta_kernel<<<(B * T) / 256, 256, 0, stream>>>(out_alpha, sexp, out_beta);
    cv_kernel<<<dim3(D / 256, B), 256, 0, stream>>>(value, out_beta, out_cv);
}
